// LSTMAutoEncoder_51067161150025
// MI455X (gfx1250) — compile-verified, matrix operand order checked by reading, not run on hardware
//
#include <hip/hip_runtime.h>

// ---------------------------------------------------------------------------
// LSTM auto-encoder for MI455X (gfx1250, wave32, WMMA 16x16x32 bf16).
//   * Precompute x@Wih^T + b ("gx") for both encoder layers with a big
//     parallel WMMA GEMM (recurrence-free part of the LSTM).
//   * Persistent recurrent kernels: 16 batch rows per block, cell state in
//     registers (C/D accumulator layout), h double-buffered in LDS (bf16).
//     Whh preloaded ONCE into LDS via global_load_async_to_lds_b128 (CDNA5
//     async copy, ASYNCcnt) -- recurrent steps then only touch LDS + gx.
//   * Fused attention: q.k^T WMMA -> shfl_xor softmax -> LDS relayout ->
//     probs.v WMMA against pre-transposed v^T[B,H,S].
//   * Decoder: sequential persistent kernel, 2 LSTM cells + fc per step;
//     W0+fcW LDS-resident (async preload), W1 streamed from L2 (192MB).
// ---------------------------------------------------------------------------

#define BB 512
#define SS 256
#define DD 64
#define HH 128
#define G4 512   // 4*H

typedef __attribute__((ext_vector_type(16))) __bf16       v16bf;
typedef __attribute__((ext_vector_type(8)))  float        v8f;
typedef __attribute__((ext_vector_type(4)))  unsigned int v4u;

// ---- scalar helpers -------------------------------------------------------
__device__ __forceinline__ unsigned short f2bf(float f) {
    unsigned int x = __float_as_uint(f);
    unsigned int r = x + 0x7fffu + ((x >> 16) & 1u);   // round-to-nearest-even
    return (unsigned short)(r >> 16);
}
__device__ __forceinline__ float bf2f(unsigned short u) {
    return __uint_as_float(((unsigned int)u) << 16);
}
__device__ __forceinline__ float sigmoid_f(float x) {
    return 1.0f / (1.0f + __expf(-x));
}
__device__ __forceinline__ float tanh_f(float x) {
    float e = __expf(2.0f * x);
    return (e - 1.0f) / (e + 1.0f);
}

// ---- CDNA5 async global->LDS copy (ASYNCcnt path) -------------------------
// 16 bytes per lane per issue; LDS dest address = low 32 bits of the generic
// pointer (addrspace(3) offset), global source in a VGPR pair.
__device__ __forceinline__ void async_copy_b128(void* lds_dst, const void* gsrc) {
    unsigned off = (unsigned)(size_t)lds_dst;
    asm volatile("global_load_async_to_lds_b128 %0, %1, off"
                 :: "v"(off), "v"(gsrc)
                 : "memory");
}
__device__ __forceinline__ void async_wait0() {
    asm volatile("s_wait_asynccnt 0" ::: "memory");
}

// ---- WMMA wrappers --------------------------------------------------------
__device__ __forceinline__ v8f wmma_bf16(v16bf a, v16bf b, v8f c) {
    return __builtin_amdgcn_wmma_f32_16x16x32_bf16(false, a, false, b,
                                                   (short)0, c, false, false);
}

// Load a 16x32 bf16 fragment (A: rows x K, or B: cols x K when weights are
// stored [N,K] row-major). Per ISA 16-bit A layout: lanes 0-15 hold rows 0-15
// with K = {0..7, 16..23}; lanes 16-31 hold the same rows with K = {8..15,
// 24..31}. Each lane = two contiguous 16B loads.
__device__ __forceinline__ v16bf load_frag_bf(const unsigned short* p0,
                                              int rstride, int lane) {
    int r  = lane & 15;
    int hf = (lane >> 4) << 3;
    const unsigned short* q = p0 + (size_t)r * (size_t)rstride + hf;
    union { v4u x[2]; v16bf v; } u;
    u.x[0] = *(const v4u*)(q);
    u.x[1] = *(const v4u*)(q + 16);
    return u.v;
}

// LSTM gate fusion on C/D-layout accumulators. acc[0..1]=i, [2..3]=f,
// [4..5]=g, [6..7]=o (two 16x16 tiles per gate covering a 16x32 hidden slice).
__device__ __forceinline__ void lstm_update(v8f* acc, float* cst,
                                            unsigned short (*hnxt)[HH],
                                            int jbase, int lane,
                                            unsigned short* yrow,
                                            size_t ystride) {
#pragma unroll
    for (int t = 0; t < 2; ++t) {
#pragma unroll
        for (int v = 0; v < 8; ++v) {
            float iv = sigmoid_f(acc[0 + t][v]);
            float fv = sigmoid_f(acc[2 + t][v]);
            float gv = tanh_f(acc[4 + t][v]);
            float ov = sigmoid_f(acc[6 + t][v]);
            float c2 = fv * cst[t * 8 + v] + iv * gv;
            cst[t * 8 + v] = c2;
            float h2 = ov * tanh_f(c2);
            int M = ((lane >> 4) << 3) + v;
            int N = jbase + t * 16 + (lane & 15);
            unsigned short hb = f2bf(h2);
            hnxt[M][N] = hb;
            if (yrow) yrow[(size_t)M * ystride + N] = hb;
        }
    }
}

// ---------------------------------------------------------------------------
// Small utility kernels
// ---------------------------------------------------------------------------
__global__ void f32_to_bf16_kernel(const float* __restrict__ src,
                                   unsigned short* __restrict__ dst, size_t n) {
    for (size_t i = blockIdx.x * (size_t)blockDim.x + threadIdx.x; i < n;
         i += (size_t)gridDim.x * blockDim.x)
        dst[i] = f2bf(src[i]);
}

__global__ void add_bias_kernel(const float* __restrict__ a,
                                const float* __restrict__ b,
                                float* __restrict__ o, int n) {
    int i = blockIdx.x * blockDim.x + threadIdx.x;
    if (i < n) o[i] = a[i] + b[i];
}

// dst[n, 0..Ka) = A[n,:], dst[n, Ka..Ka+Kb) = B[n,:]  (f32 -> bf16)
__global__ void pack2_bf16_kernel(const float* __restrict__ A, int Ka,
                                  const float* __restrict__ Bm, int Kb,
                                  unsigned short* __restrict__ dst, int N) {
    int Kt = Ka + Kb;
    size_t tot = (size_t)N * Kt;
    for (size_t i = blockIdx.x * (size_t)blockDim.x + threadIdx.x; i < tot;
         i += (size_t)gridDim.x * blockDim.x) {
        int kk = (int)(i % Kt);
        int n  = (int)(i / Kt);
        float v = (kk < Ka) ? A[(size_t)n * Ka + kk]
                            : Bm[(size_t)n * Kb + (kk - Ka)];
        dst[i] = f2bf(v);
    }
}

// vT[b,h,s] = v[s,b,h]   (bf16 -> bf16)
__global__ void transpose_v_kernel(const unsigned short* __restrict__ v,
                                   unsigned short* __restrict__ vT) {
    size_t tot = (size_t)BB * HH * SS;
    for (size_t i = blockIdx.x * (size_t)blockDim.x + threadIdx.x; i < tot;
         i += (size_t)gridDim.x * blockDim.x) {
        int s = (int)(i % SS);
        size_t r = i / SS;
        int h = (int)(r % HH);
        int b = (int)(r / HH);
        vT[i] = v[((size_t)s * BB + b) * HH + h];
    }
}

// dec h init: d{0,1}[b,h] = h{0,1}_enc[b,h] + sum_s attn[b,s,h] * Pw[l,s]
__global__ void proj_combine_kernel(const float* __restrict__ attn,
                                    const float* __restrict__ Pw,
                                    const float* __restrict__ h0e,
                                    const float* __restrict__ h1e,
                                    float* __restrict__ d0,
                                    float* __restrict__ d1) {
    int idx = blockIdx.x * blockDim.x + threadIdx.x;
    if (idx >= BB * HH) return;
    int b = idx / HH, h = idx % HH;
    float s0 = 0.f, s1 = 0.f;
    for (int s = 0; s < SS; ++s) {
        float a = attn[((size_t)b * SS + s) * HH + h];
        s0 += a * Pw[s];
        s1 += a * Pw[SS + s];
    }
    d0[idx] = h0e[idx] + s0;
    d1[idx] = h1e[idx] + s1;
}

// ---------------------------------------------------------------------------
// Generic WMMA GEMM:  C[m,n] = sum_k A[m,k]*W[n,k] + bias[n]
//   A bf16 [M,K] (row stride lda), W bf16 [N,K], C f32 and/or bf16.
//   Output row remap (for gx[s,b,:] from x[b,s,:] ordering) via swap_sb.
// Block: 256 threads = 8 waves: wave (rt = wid>>2 in 0..1, jt = wid&3),
// tile = 32 rows x 128 cols. grid = (M/32, N/128).
// ---------------------------------------------------------------------------
__global__ void gemm_bf16_kernel(const unsigned short* __restrict__ A, int lda,
                                 const unsigned short* __restrict__ W, int K,
                                 const float* __restrict__ bias,
                                 float* __restrict__ Cf,
                                 unsigned short* __restrict__ Cb, int ldc,
                                 int swap_sb, int s_mask, int s_shift,
                                 int b_dim) {
    int tid = threadIdx.x, wid = tid >> 5, lane = tid & 31;
    int rt = wid >> 2, jt = wid & 3;
    int mbase = blockIdx.x * 32 + rt * 16;
    int nb0   = blockIdx.y * 128 + jt * 32;

    v8f acc[2];
#pragma unroll
    for (int t = 0; t < 2; ++t) {
        float bv = bias ? bias[nb0 + t * 16 + (lane & 15)] : 0.0f;
#pragma unroll
        for (int v = 0; v < 8; ++v) acc[t][v] = bv;
    }

    int K32 = K >> 5;
    const unsigned short* Ap = A + (size_t)mbase * lda;
    for (int kc = 0; kc < K32; ++kc) {
        int kb = kc << 5;
        __builtin_prefetch(Ap + kb + 32, 0, 1);
        v16bf af = load_frag_bf(Ap + kb, lda, lane);
#pragma unroll
        for (int t = 0; t < 2; ++t) {
            const unsigned short* wp = W + (size_t)(nb0 + t * 16) * K + kb;
            v16bf bf = load_frag_bf(wp, K, lane);
            acc[t] = wmma_bf16(af, bf, acc[t]);
        }
    }

#pragma unroll
    for (int t = 0; t < 2; ++t) {
#pragma unroll
        for (int v = 0; v < 8; ++v) {
            int m_e = mbase + ((lane >> 4) << 3) + v;
            int n_e = nb0 + t * 16 + (lane & 15);
            int orow = swap_sb ? ((m_e & s_mask) * b_dim + (m_e >> s_shift))
                               : m_e;
            float val = acc[t][v];
            if (Cf) Cf[(size_t)orow * ldc + n_e] = val;
            if (Cb) Cb[(size_t)orow * ldc + n_e] = f2bf(val);
        }
    }
}

// ---------------------------------------------------------------------------
// Persistent encoder LSTM layer. gx[s,b,0:512] holds x@Wih^T + b.
// Block = 16 batch rows, 128 threads = 4 waves, wave w owns hidden cols
// w*32..w*32+31 (=> 8 gate tiles: 4 gate-groups x 2 tiles).  Per step:
//   gates = gx[s] + h_prev @ Whh^T   (4 K-chunks x 8 tiles = 32 WMMAs)
// Whh resides in LDS (async preload); h double-buffered in LDS; cell state
// in registers.  Next step's gx slice is prefetched during the current step.
// ---------------------------------------------------------------------------
__global__ void lstm_recurrent_kernel(const unsigned short* __restrict__ Whh,
                                      const float* __restrict__ gx,
                                      unsigned short* __restrict__ y,
                                      float* __restrict__ hN,
                                      float* __restrict__ cN) {
    __shared__ unsigned short wlds[G4 * HH];      // 128 KB
    __shared__ unsigned short hbuf[2][16][HH];    // 8 KB
    int bbase = blockIdx.x * 16;
    int tid = threadIdx.x, wid = tid >> 5, lane = tid & 31;
    int jbase = wid * 32;

    // async preload Whh -> LDS (each lane copies 16B per issue)
    for (int i = tid * 8; i < G4 * HH; i += 128 * 8)
        async_copy_b128(&wlds[i], Whh + i);

    float cst[16];
#pragma unroll
    for (int i = 0; i < 16; ++i) cst[i] = 0.f;
    {
        unsigned short* hz = &hbuf[0][0][0];
        for (int i = tid; i < 2 * 16 * HH; i += 128) hz[i] = 0;
    }
    async_wait0();
    __syncthreads();

    int cur = 0;
    for (int s = 0; s < SS; ++s) {
        const float* gxp = gx + ((size_t)s * BB + bbase) * G4;
        if (s + 1 < SS) {
            const char* nx =
                (const char*)(gx + ((size_t)(s + 1) * BB + bbase) * G4);
            __builtin_prefetch(nx + tid * 256, 0, 1);        // 32 KB slice
        }
        v8f acc[8];
#pragma unroll
        for (int g = 0; g < 4; ++g) {
#pragma unroll
            for (int t = 0; t < 2; ++t) {
                int N  = g * HH + jbase + t * 16 + (lane & 15);
                int mo = (lane >> 4) << 3;
                v8f a;
#pragma unroll
                for (int v = 0; v < 8; ++v)
                    a[v] = gxp[(size_t)(mo + v) * G4 + N];
                acc[g * 2 + t] = a;
            }
        }
#pragma unroll
        for (int kc = 0; kc < 4; ++kc) {
            int kb = kc << 5;
            v16bf af = load_frag_bf(&hbuf[cur][0][0] + kb, HH, lane);
#pragma unroll
            for (int g = 0; g < 4; ++g) {
#pragma unroll
                for (int t = 0; t < 2; ++t) {
                    int nb = g * HH + jbase + t * 16;
                    v16bf bf = load_frag_bf(&wlds[(size_t)nb * HH + kb], HH, lane);
                    acc[g * 2 + t] = wmma_bf16(af, bf, acc[g * 2 + t]);
                }
            }
        }
        lstm_update(acc, cst, hbuf[cur ^ 1], jbase, lane,
                    y + ((size_t)s * BB + bbase) * HH, (size_t)HH);
        __syncthreads();
        cur ^= 1;
    }

    // final h, c (f32)
#pragma unroll
    for (int t = 0; t < 2; ++t) {
#pragma unroll
        for (int v = 0; v < 8; ++v) {
            int M = ((lane >> 4) << 3) + v;
            int N = jbase + t * 16 + (lane & 15);
            hN[(size_t)(bbase + M) * HH + N] = bf2f(hbuf[cur][M][N]);
            cN[(size_t)(bbase + M) * HH + N] = cst[t * 8 + v];
        }
    }
}

// ---------------------------------------------------------------------------
// Fused self-attention for one (batch, 64-row) tile.
// Block = 128 threads = 4 waves; wave owns 16 query rows.
// q,k bf16 [S,B,H]; vT bf16 [B,H,S]; out f32 [B,S,H].
// ---------------------------------------------------------------------------
__global__ void attention_kernel(const unsigned short* __restrict__ q,
                                 const unsigned short* __restrict__ k,
                                 const unsigned short* __restrict__ vT,
                                 float* __restrict__ attn_out) {
    __shared__ unsigned short probs[64][SS];
    int b = blockIdx.x, st = blockIdx.y;
    int tid = threadIdx.x, wid = tid >> 5, lane = tid & 31;
    int srow = st * 64 + wid * 16;
    const int rs = BB * HH;  // row stride along s

    // cache q fragments (K = H = 128 -> 4 chunks)
    const unsigned short* qp = q + (size_t)srow * rs + (size_t)b * HH;
    v16bf qf[4];
#pragma unroll
    for (int kc = 0; kc < 4; ++kc)
        qf[kc] = load_frag_bf(qp + kc * 32, rs, lane);

    // scores = q @ k^T  (16 tiles of 16x16 across all S keys)
    v8f sc[16];
#pragma unroll
    for (int nt = 0; nt < 16; ++nt) {
        v8f a;
#pragma unroll
        for (int v = 0; v < 8; ++v) a[v] = 0.f;
#pragma unroll
        for (int kc = 0; kc < 4; ++kc) {
            const unsigned short* kp =
                k + (size_t)(nt * 16) * rs + (size_t)b * HH + kc * 32;
            v16bf bf = load_frag_bf(kp, rs, lane);
            a = wmma_bf16(qf[kc], bf, a);
        }
        sc[nt] = a;
    }

    // row softmax: each C/D row lives in one 16-lane half of the wave.
#pragma unroll
    for (int v = 0; v < 8; ++v) {
        float m = -3.0e38f;
#pragma unroll
        for (int nt = 0; nt < 16; ++nt) m = fmaxf(m, sc[nt][v]);
#pragma unroll
        for (int off = 1; off < 16; off <<= 1)
            m = fmaxf(m, __shfl_xor(m, off, 32));
        float sum = 0.f;
#pragma unroll
        for (int nt = 0; nt < 16; ++nt) {
            float e = __expf(sc[nt][v] - m);
            sc[nt][v] = e;
            sum += e;
        }
#pragma unroll
        for (int off = 1; off < 16; off <<= 1)
            sum += __shfl_xor(sum, off, 32);
        float inv = 1.0f / sum;
        int M = wid * 16 + ((lane >> 4) << 3) + v;
#pragma unroll
        for (int nt = 0; nt < 16; ++nt)
            probs[M][nt * 16 + (lane & 15)] = f2bf(sc[nt][v] * inv);
    }
    __syncthreads();

    // out = probs @ v   (K = S = 256 -> 8 chunks; B-frags contiguous via vT)
    v8f oacc[8];
#pragma unroll
    for (int nt = 0; nt < 8; ++nt)
#pragma unroll
        for (int v = 0; v < 8; ++v) oacc[nt][v] = 0.f;

    const unsigned short* vb = vT + (size_t)b * HH * SS;
#pragma unroll
    for (int kc = 0; kc < 8; ++kc) {
        int kb = kc << 5;
        v16bf af = load_frag_bf(&probs[wid * 16][0] + kb, SS, lane);
#pragma unroll
        for (int nt = 0; nt < 8; ++nt) {
            v16bf bf = load_frag_bf(vb + (size_t)(nt * 16) * SS + kb, SS, lane);
            oacc[nt] = wmma_bf16(af, bf, oacc[nt]);
        }
    }
#pragma unroll
    for (int nt = 0; nt < 8; ++nt) {
#pragma unroll
        for (int v = 0; v < 8; ++v) {
            int M = srow + ((lane >> 4) << 3) + v;
            int N = nt * 16 + (lane & 15);
            attn_out[((size_t)b * SS + M) * HH + N] = oacc[nt][v];
        }
    }
}

// ---------------------------------------------------------------------------
// Persistent autoregressive decoder: 2 LSTM cells + fc projection per step,
// fc output fed back as next input through LDS. Block = 16 batch rows,
// 128 threads = 4 waves. W0 ([512,192]) + fcW LDS-resident (async preload,
// ~208KB of the 320KB WGP LDS); W1 ([512,256]) streamed from L2.
// ---------------------------------------------------------------------------
__global__ void decoder_kernel(const unsigned short* __restrict__ W0,  // [512,192]
                               const unsigned short* __restrict__ W1,  // [512,256]
                               const unsigned short* __restrict__ fcW, // [64,128]
                               const float* __restrict__ b0,
                               const float* __restrict__ b1,
                               const float* __restrict__ fcb,
                               const float* __restrict__ h0i,
                               const float* __restrict__ c0i,
                               const float* __restrict__ h1i,
                               const float* __restrict__ c1i,
                               float* __restrict__ out) {
    __shared__ unsigned short w0lds[G4 * 192];    // 192 KB
    __shared__ unsigned short fclds[DD * HH];     // 16 KB
    __shared__ unsigned short xin[2][16][DD];
    __shared__ unsigned short h0b[2][16][HH];
    __shared__ unsigned short h1b[2][16][HH];
    int bbase = blockIdx.x * 16;
    int tid = threadIdx.x, wid = tid >> 5, lane = tid & 31;
    int jbase = wid * 32;

    // async preload decoder layer-0 weights + fc weights into LDS
    for (int i = tid * 8; i < G4 * 192; i += 128 * 8)
        async_copy_b128(&w0lds[i], W0 + i);
    for (int i = tid * 8; i < DD * HH; i += 128 * 8)
        async_copy_b128(&fclds[i], fcW + i);

    for (int i = tid; i < 16 * HH; i += 128) {
        int r = i >> 7, cc = i & 127;
        h0b[0][r][cc] = f2bf(h0i[(size_t)(bbase + r) * HH + cc]);
        h1b[0][r][cc] = f2bf(h1i[(size_t)(bbase + r) * HH + cc]);
    }
    for (int i = tid; i < 16 * DD; i += 128) xin[0][i >> 6][i & 63] = 0;

    float c0[16], c1[16];
#pragma unroll
    for (int t = 0; t < 2; ++t) {
#pragma unroll
        for (int v = 0; v < 8; ++v) {
            int M = ((lane >> 4) << 3) + v;
            int N = jbase + t * 16 + (lane & 15);
            c0[t * 8 + v] = c0i[(size_t)(bbase + M) * HH + N];
            c1[t * 8 + v] = c1i[(size_t)(bbase + M) * HH + N];
        }
    }
    async_wait0();
    __syncthreads();

    int cur = 0;
    for (int s = 0; s < SS; ++s) {
        int nxt = cur ^ 1;
        v8f acc[8];

        // ---- layer 0: gates = [x | h0] @ W0^T + b0  (K = 64+128 = 192)
#pragma unroll
        for (int g = 0; g < 4; ++g)
#pragma unroll
            for (int t = 0; t < 2; ++t) {
                float bv = b0[g * HH + jbase + t * 16 + (lane & 15)];
#pragma unroll
                for (int v = 0; v < 8; ++v) acc[g * 2 + t][v] = bv;
            }
#pragma unroll
        for (int kc = 0; kc < 6; ++kc) {
            int kb = kc << 5;
            v16bf af = (kb < DD)
                           ? load_frag_bf(&xin[cur][0][0] + kb, DD, lane)
                           : load_frag_bf(&h0b[cur][0][0] + (kb - DD), HH, lane);
#pragma unroll
            for (int g = 0; g < 4; ++g)
#pragma unroll
                for (int t = 0; t < 2; ++t) {
                    int nb = g * HH + jbase + t * 16;
                    v16bf bf = load_frag_bf(&w0lds[(size_t)nb * 192 + kb], 192, lane);
                    acc[g * 2 + t] = wmma_bf16(af, bf, acc[g * 2 + t]);
                }
        }
        lstm_update(acc, c0, h0b[nxt], jbase, lane, nullptr, 0);
        __syncthreads();

        // ---- layer 1: gates = [h0_new | h1] @ W1^T + b1  (K = 256, L2-hot)
#pragma unroll
        for (int g = 0; g < 4; ++g)
#pragma unroll
            for (int t = 0; t < 2; ++t) {
                float bv = b1[g * HH + jbase + t * 16 + (lane & 15)];
#pragma unroll
                for (int v = 0; v < 8; ++v) acc[g * 2 + t][v] = bv;
            }
#pragma unroll
        for (int kc = 0; kc < 8; ++kc) {
            int kb = kc << 5;
            v16bf af = (kb < HH)
                           ? load_frag_bf(&h0b[nxt][0][0] + kb, HH, lane)
                           : load_frag_bf(&h1b[cur][0][0] + (kb - HH), HH, lane);
#pragma unroll
            for (int g = 0; g < 4; ++g)
#pragma unroll
                for (int t = 0; t < 2; ++t) {
                    int nb = g * HH + jbase + t * 16;
                    v16bf bf = load_frag_bf(W1 + (size_t)nb * 256 + kb, 256, lane);
                    acc[g * 2 + t] = wmma_bf16(af, bf, acc[g * 2 + t]);
                }
        }
        lstm_update(acc, c1, h1b[nxt], jbase, lane, nullptr, 0);
        __syncthreads();

        // ---- fc: pred = h1_new @ fcW^T + fcb  (each wave: one 16x16 tile)
        {
            int nf = wid * 16;
            v8f a;
            float bv = fcb[nf + (lane & 15)];
#pragma unroll
            for (int v = 0; v < 8; ++v) a[v] = bv;
#pragma unroll
            for (int kc = 0; kc < 4; ++kc) {
                int kb = kc << 5;
                v16bf af = load_frag_bf(&h1b[nxt][0][0] + kb, HH, lane);
                v16bf bf = load_frag_bf(&fclds[(size_t)nf * HH + kb], HH, lane);
                a = wmma_bf16(af, bf, a);
            }
#pragma unroll
            for (int v = 0; v < 8; ++v) {
                int M = ((lane >> 4) << 3) + v;
                int N = nf + (lane & 15);
                out[((size_t)(bbase + M) * SS + s) * DD + N] = a[v];
                xin[nxt][M][N] = f2bf(a[v]);
            }
        }
        __syncthreads();
        cur = nxt;
    }
}

// ---------------------------------------------------------------------------
// Host orchestration
// ---------------------------------------------------------------------------
extern "C" void kernel_launch(void* const* d_in, const int* in_sizes, int n_in,
                              void* d_out, int out_size, void* d_ws,
                              size_t ws_size, hipStream_t stream) {
    const float* x     = (const float*)d_in[0];
    const float* eWih0 = (const float*)d_in[1];
    const float* eWhh0 = (const float*)d_in[2];
    const float* ebih0 = (const float*)d_in[3];
    const float* ebhh0 = (const float*)d_in[4];
    const float* eWih1 = (const float*)d_in[5];
    const float* eWhh1 = (const float*)d_in[6];
    const float* ebih1 = (const float*)d_in[7];
    const float* ebhh1 = (const float*)d_in[8];
    const float* dWih0 = (const float*)d_in[9];
    const float* dWhh0 = (const float*)d_in[10];
    const float* dbih0 = (const float*)d_in[11];
    const float* dbhh0 = (const float*)d_in[12];
    const float* dWih1 = (const float*)d_in[13];
    const float* dWhh1 = (const float*)d_in[14];
    const float* dbih1 = (const float*)d_in[15];
    const float* dbhh1 = (const float*)d_in[16];
    const float* fcW   = (const float*)d_in[17];
    const float* fcb   = (const float*)d_in[18];
    const float* Qw    = (const float*)d_in[19];
    const float* Kw    = (const float*)d_in[20];
    const float* Vw    = (const float*)d_in[21];
    const float* Pw    = (const float*)d_in[22];
    float* out = (float*)d_out;

    char* ws = (char*)d_ws;
    size_t off = 0;
    auto alloc = [&](size_t bytes) -> char* {
        char* p = ws + off;
        off = (off + bytes + 255) & ~(size_t)255;
        return p;
    };

    unsigned short* xb   = (unsigned short*)alloc((size_t)BB * SS * DD * 2);
    unsigned short* y0   = (unsigned short*)alloc((size_t)SS * BB * HH * 2);
    unsigned short* y1   = (unsigned short*)alloc((size_t)SS * BB * HH * 2);
    unsigned short* qb   = (unsigned short*)alloc((size_t)SS * BB * HH * 2);
    unsigned short* kb   = (unsigned short*)alloc((size_t)SS * BB * HH * 2);
    unsigned short* vb   = (unsigned short*)alloc((size_t)SS * BB * HH * 2);
    unsigned short* vTb  = (unsigned short*)alloc((size_t)BB * HH * SS * 2);
    float*          gx   = (float*)alloc((size_t)SS * BB * G4 * 4);
    float*          attn = (float*)alloc((size_t)BB * SS * HH * 4);
    float* h0e = (float*)alloc((size_t)BB * HH * 4);
    float* c0e = (float*)alloc((size_t)BB * HH * 4);
    float* h1e = (float*)alloc((size_t)BB * HH * 4);
    float* c1e = (float*)alloc((size_t)BB * HH * 4);
    float* dh0 = (float*)alloc((size_t)BB * HH * 4);
    float* dh1 = (float*)alloc((size_t)BB * HH * 4);
    unsigned short* Wih0b = (unsigned short*)alloc((size_t)G4 * DD * 2);
    unsigned short* Whh0b = (unsigned short*)alloc((size_t)G4 * HH * 2);
    unsigned short* Wih1b = (unsigned short*)alloc((size_t)G4 * HH * 2);
    unsigned short* Whh1b = (unsigned short*)alloc((size_t)G4 * HH * 2);
    unsigned short* Qwb   = (unsigned short*)alloc((size_t)HH * HH * 2);
    unsigned short* Kwb   = (unsigned short*)alloc((size_t)HH * HH * 2);
    unsigned short* Vwb   = (unsigned short*)alloc((size_t)HH * HH * 2);
    unsigned short* W0d   = (unsigned short*)alloc((size_t)G4 * 192 * 2);
    unsigned short* W1d   = (unsigned short*)alloc((size_t)G4 * 256 * 2);
    unsigned short* fcWb  = (unsigned short*)alloc((size_t)DD * HH * 2);
    float* be0 = (float*)alloc(G4 * 4);
    float* be1 = (float*)alloc(G4 * 4);
    float* bd0 = (float*)alloc(G4 * 4);
    float* bd1 = (float*)alloc(G4 * 4);

    auto cvt = [&](const float* s, unsigned short* d, size_t n) {
        int blocks = (int)((n + 255) / 256);
        if (blocks > 32768) blocks = 32768;
        f32_to_bf16_kernel<<<blocks, 256, 0, stream>>>(s, d, n);
    };

    // weight / input conversion
    cvt(x, xb, (size_t)BB * SS * DD);
    cvt(eWih0, Wih0b, (size_t)G4 * DD);
    cvt(eWhh0, Whh0b, (size_t)G4 * HH);
    cvt(eWih1, Wih1b, (size_t)G4 * HH);
    cvt(eWhh1, Whh1b, (size_t)G4 * HH);
    cvt(Qw, Qwb, (size_t)HH * HH);
    cvt(Kw, Kwb, (size_t)HH * HH);
    cvt(Vw, Vwb, (size_t)HH * HH);
    cvt(fcW, fcWb, (size_t)DD * HH);
    pack2_bf16_kernel<<<768, 256, 0, stream>>>(dWih0, DD, dWhh0, HH, W0d, G4);
    pack2_bf16_kernel<<<1024, 256, 0, stream>>>(dWih1, HH, dWhh1, HH, W1d, G4);
    add_bias_kernel<<<2, 256, 0, stream>>>(ebih0, ebhh0, be0, G4);
    add_bias_kernel<<<2, 256, 0, stream>>>(ebih1, ebhh1, be1, G4);
    add_bias_kernel<<<2, 256, 0, stream>>>(dbih0, dbhh0, bd0, G4);
    add_bias_kernel<<<2, 256, 0, stream>>>(dbih1, dbhh1, bd1, G4);

    const int M = BB * SS;

    // encoder layer 0: gx[s,b,:] = x[b,s,:] @ Wih0^T + b0  (output remapped)
    gemm_bf16_kernel<<<dim3(M / 32, G4 / 128), 256, 0, stream>>>(
        xb, DD, Wih0b, DD, be0, gx, nullptr, G4, 1, SS - 1, 8, BB);
    lstm_recurrent_kernel<<<BB / 16, 128, 0, stream>>>(Whh0b, gx, y0, h0e, c0e);

    // encoder layer 1
    gemm_bf16_kernel<<<dim3(M / 32, G4 / 128), 256, 0, stream>>>(
        y0, HH, Wih1b, HH, be1, gx, nullptr, G4, 0, 0, 0, 0);
    lstm_recurrent_kernel<<<BB / 16, 128, 0, stream>>>(Whh1b, gx, y1, h1e, c1e);

    // q, k, v projections (bf16 outputs, no bias)
    gemm_bf16_kernel<<<dim3(M / 32, 1), 256, 0, stream>>>(
        y1, HH, Qwb, HH, nullptr, nullptr, qb, HH, 0, 0, 0, 0);
    gemm_bf16_kernel<<<dim3(M / 32, 1), 256, 0, stream>>>(
        y1, HH, Kwb, HH, nullptr, nullptr, kb, HH, 0, 0, 0, 0);
    gemm_bf16_kernel<<<dim3(M / 32, 1), 256, 0, stream>>>(
        y1, HH, Vwb, HH, nullptr, nullptr, vb, HH, 0, 0, 0, 0);

    transpose_v_kernel<<<16384, 256, 0, stream>>>(vb, vTb);
    attention_kernel<<<dim3(BB, SS / 64), 128, 0, stream>>>(qb, kb, vTb, attn);
    proj_combine_kernel<<<(BB * HH + 255) / 256, 256, 0, stream>>>(
        attn, Pw, h0e, h1e, dh0, dh1);

    // autoregressive decoder
    decoder_kernel<<<BB / 16, 128, 0, stream>>>(W0d, W1d, fcWb, bd0, bd1, fcb,
                                                dh0, c0e, dh1, c1e, out);
    (void)in_sizes; (void)n_in; (void)out_size; (void)ws_size;
}
